// _GraphAttentionLayer_47210280518129
// MI455X (gfx1250) — matrix operand order repaired and checked by reading, not compiled, not run on hardware
//
#include <hip/hip_runtime.h>

#define N 8192
#define IN_F 512
#define OUT_F 128

typedef __attribute__((ext_vector_type(2)))  float  v2f;
typedef __attribute__((ext_vector_type(8)))  float  v8f;
typedef __attribute__((ext_vector_type(4)))  int    v4i;
typedef __attribute__((ext_vector_type(16))) __bf16 v16bf;

// ---------------------------------------------------------------------------
// Kernel 1: Wh = h @ W.T   [8192x512] x [512x128] -> [8192x128], exact fp32
// via V_WMMA_F32_16X16X4_F32. One wave per 16x16 output tile.
// A 16x4 f32 layout: lane L (r=L&15, h=L>>4): VGPR v = h[ibase+r][k + v + 2h]
// B 4x16  f32 layout: VGPR v = W[cbase+r][k + v + 2h]  (B[k][n] = W.T[k][n])
// Each lane loads exactly its two consecutive floats -> one b64 per operand.
// ---------------------------------------------------------------------------
__global__ void __launch_bounds__(128)
gat_wh_kernel(const float* __restrict__ hmat,
              const float* __restrict__ W,
              float* __restrict__ Wh) {
  const int lane = threadIdx.x & 31;
  const int wave = threadIdx.x >> 5;
  const int tile = blockIdx.x * 4 + wave;     // 4096 tiles = 512 x 8
  const int ibase = (tile >> 3) * 16;
  const int cbase = (tile & 7) * 16;
  const int r  = lane & 15;
  const int hh = lane >> 4;

  const float* hrow = hmat + (size_t)(ibase + r) * IN_F + 2 * hh;
  const float* wrow = W    + (size_t)(cbase + r) * IN_F + 2 * hh;

  v8f acc = {};
  for (int k = 0; k < IN_F; k += 4) {
    v2f A = *(const v2f*)(hrow + k);
    v2f B = *(const v2f*)(wrow + k);
    acc = __builtin_amdgcn_wmma_f32_16x16x4_f32(false, A, false, B,
                                                (short)0, acc, false, false);
  }
  // C/D layout: VGPR v -> row v + 8*hh, col = cbase + r
#pragma unroll
  for (int v = 0; v < 8; ++v)
    Wh[(size_t)(ibase + v + 8 * hh) * OUT_F + cbase + r] = acc[v];
}

// ---------------------------------------------------------------------------
// Kernel 2: f1[i] = Wh[i,:].a1 ; f2[i] = Wh[i,:].a2 ; WhT (bf16, transposed)
// ---------------------------------------------------------------------------
__global__ void __launch_bounds__(256)
gat_fvec_kernel(const float* __restrict__ Wh,
                const float* __restrict__ avec,
                float* __restrict__ f1,
                float* __restrict__ f2,
                unsigned short* __restrict__ WhT) {
  const int i = blockIdx.x * blockDim.x + threadIdx.x;
  if (i >= N) return;
  float s1 = 0.0f, s2 = 0.0f;
  for (int c = 0; c < OUT_F; ++c) {
    float w = Wh[(size_t)i * OUT_F + c];
    s1 = fmaf(w, avec[c], s1);
    s2 = fmaf(w, avec[OUT_F + c], s2);
    __bf16 b = (__bf16)w;
    WhT[(size_t)c * N + i] = __builtin_bit_cast(unsigned short, b);
  }
  f1[i] = s1;
  f2[i] = s2;
}

// ---------------------------------------------------------------------------
// Kernel 3: fused masked-softmax @ Wh (flash-GAT), one pass over adj.
// Block = 16 rows, 4 waves split the j dimension; bf16 WMMA 16x16x32,
// f32 accumulate; deterministic LDS tree reduction (no float atomics).
// adj is streamed once (256 MB) -> non-temporal loads keep L2 for WhT/f2.
// ---------------------------------------------------------------------------
__global__ void __launch_bounds__(128)
gat_attn_kernel(const int* __restrict__ adj,
                const unsigned short* __restrict__ WhTu,
                const float* __restrict__ f1,
                const float* __restrict__ f2,
                float* __restrict__ out) {
  __shared__ float lds_acc[4][16 * OUT_F];
  __shared__ float lds_rs[4][2][16];

  const __bf16* WhT = (const __bf16*)WhTu;
  const int lane = threadIdx.x & 31;
  const int wave = threadIdx.x >> 5;
  const int r  = lane & 15;
  const int hh = lane >> 4;
  const int ibase = blockIdx.x * 16;
  const int row = ibase + r;
  const float f1r = f1[row];

  v8f zero = {};
  v8f acc[8];
#pragma unroll
  for (int t = 0; t < 8; ++t) acc[t] = zero;
  float rs = 0.0f;

  const int* adjrow = adj + (size_t)row * N;

  for (int jb = wave * 32; jb < N; jb += 128) {
    // adj / f2 for this lane's 16 K slots: k = e + 16h
    const v4i* ap = (const v4i*)(adjrow + jb + 16 * hh);
    v4i A0 = __builtin_nontemporal_load(ap + 0);
    v4i A1 = __builtin_nontemporal_load(ap + 1);
    v4i A2 = __builtin_nontemporal_load(ap + 2);
    v4i A3 = __builtin_nontemporal_load(ap + 3);
    const float4* fp = (const float4*)(f2 + jb + 16 * hh);
    float4 G0 = fp[0], G1 = fp[1], G2 = fp[2], G3 = fp[3];

    int   am[16] = {A0[0], A0[1], A0[2], A0[3], A1[0], A1[1], A1[2], A1[3],
                    A2[0], A2[1], A2[2], A2[3], A3[0], A3[1], A3[2], A3[3]};
    float fv[16] = {G0.x, G0.y, G0.z, G0.w, G1.x, G1.y, G1.z, G1.w,
                    G2.x, G2.y, G2.z, G2.w, G3.x, G3.y, G3.z, G3.w};

    v16bf Am;
#pragma unroll
    for (int e = 0; e < 16; ++e) {
      float eij = f1r + fv[e];
      eij = fmaxf(eij, 0.2f * eij);            // leaky_relu(0.2)
      float pe = (am[e] > 0) ? __expf(eij) : 0.0f;
      rs += pe;                                 // all 16 belong to row r
      Am[e] = (__bf16)pe;
    }

    // B tile: 16 contiguous bf16 of WhT row (K-major) = one 32B vector load
    const __bf16* wb = WhT + jb + 16 * hh;
#pragma unroll
    for (int t = 0; t < 8; ++t) {
      v16bf Bm = *(const v16bf*)(wb + (size_t)(t * 16 + r) * N);
      acc[t] = __builtin_amdgcn_wmma_f32_16x16x32_bf16(
          false, Am, false, Bm, (short)0, acc[t], false, false);
    }
  }

  // per-wave LDS slabs: C/D layout -> m = v + 8h, c = 16t + r
#pragma unroll
  for (int t = 0; t < 8; ++t)
#pragma unroll
    for (int v = 0; v < 8; ++v)
      lds_acc[wave][(v + 8 * hh) * OUT_F + t * 16 + r] = acc[t][v];
  lds_rs[wave][hh][r] = rs;
  __syncthreads();

  // deterministic combine + normalize + elu; thread = output column
  const int c = threadIdx.x;
  for (int m = 0; m < 16; ++m) {
    float v = lds_acc[0][m * OUT_F + c] + lds_acc[1][m * OUT_F + c] +
              lds_acc[2][m * OUT_F + c] + lds_acc[3][m * OUT_F + c];
    float rsm = 0.0f;
#pragma unroll
    for (int w = 0; w < 4; ++w) rsm += lds_rs[w][0][m] + lds_rs[w][1][m];
    float hp = v / rsm;
    float o = hp > 0.0f ? hp : (__expf(hp) - 1.0f);
    __builtin_nontemporal_store(o, out + (size_t)(ibase + m) * OUT_F + c);
  }
}

// ---------------------------------------------------------------------------
extern "C" void kernel_launch(void* const* d_in, const int* in_sizes, int n_in,
                              void* d_out, int out_size, void* d_ws, size_t ws_size,
                              hipStream_t stream) {
  const float* hmat = (const float*)d_in[0];   // [8192,512] f32
  const int*   adj  = (const int*)d_in[1];     // [8192,8192] i32
  const float* W    = (const float*)d_in[2];   // [128,512] f32
  const float* avec = (const float*)d_in[3];   // [256] f32
  float* out = (float*)d_out;                  // [8192,128] f32

  char* ws = (char*)d_ws;
  float*          Wh  = (float*)ws;                                    // 4 MB
  unsigned short* WhT = (unsigned short*)(ws + (size_t)N * OUT_F * 4);  // 2 MB
  float* f1 = (float*)(ws + (size_t)N * OUT_F * 4 + (size_t)N * OUT_F * 2);
  float* f2 = f1 + N;

  // K1: 4096 tiles, 4 waves/block
  gat_wh_kernel<<<(N / 16) * (OUT_F / 16) / 4, 128, 0, stream>>>(hmat, W, Wh);
  // K2: f1/f2 + bf16 transpose
  gat_fvec_kernel<<<N / 256, 256, 0, stream>>>(Wh, avec, f1, f2, WhT);
  // K3: fused flash-GAT, one block per 16 rows
  gat_attn_kernel<<<N / 16, 128, 0, stream>>>(adj, WhT, f1, f2, out);
}
